// CustomAttentionLayer_36601711297142
// MI455X (gfx1250) — matrix operand order repaired and checked by reading, not compiled, not run on hardware
//
#include <hip/hip_runtime.h>
#include <hip/hip_bf16.h>
#include <stdint.h>

typedef __attribute__((ext_vector_type(16))) __bf16 v16bf;
typedef __attribute__((ext_vector_type(8)))  __bf16 v8bf;
typedef __attribute__((ext_vector_type(8)))  float  v8f;

#define B_   64
#define KK   4096
#define QS_  512
#define KS_  512
#define A_   256
#define VD_  512
#define NEGV -1000000.0f

// scores-kernel tiling: 128 key rows/block, 8 waves, 128 KB LDS
#define TROWS   128
#define WAVES   (TROWS / 16)
#define THREADS (WAVES * 32)

#define KSPLIT  16                // K split for the values pass

// round-to-nearest-even f32 -> bf16 (pure integer math)
static __device__ __forceinline__ unsigned short f2bf_raw(float f) {
    unsigned int u = __float_as_uint(f);
    unsigned int r = u + 0x7FFFu + ((u >> 16) & 1u);
    return (unsigned short)(r >> 16);
}

// ---------------------------------------------------------------------------
// qfeat[b][a] = queries[b]@Wq[:,a] + bq[a] + bk[a]   (both biases folded)
// ---------------------------------------------------------------------------
__global__ void k_qfeat(const float* __restrict__ q, const float* __restrict__ Wq,
                        const float* __restrict__ bq, const float* __restrict__ bk,
                        float* __restrict__ qf) {
    int b = blockIdx.x, a = threadIdx.x;
    float acc = bq[a] + bk[a];
    const float* qr = q + b * QS_;
    for (int d = 0; d < QS_; ++d) acc += qr[d] * Wq[d * A_ + a];
    qf[b * A_ + a] = acc;
}

// ---------------------------------------------------------------------------
// Pre-swizzle Wk (512x256 f32, row-major) into bf16 WMMA B-fragments.
// Fragment f = s*16 + nt covers K rows [s*32, s*32+32), cols [nt*16, nt*16+16).
// Lane L holds column N = nt*16 + L%16, K rows (L<16 ? {0..7,16..23} : {8..15,24..31}) of the
// slab, 2 bf16 per dword in K order (ISA 05_wmma 16-bit B layout).
// Memory layout: frag f -> 32 lanes x 16 bf16, lane-major (one v16bf per lane).
// ---------------------------------------------------------------------------
__global__ void k_packWk(const float* __restrict__ Wk, unsigned short* __restrict__ pk) {
    int f = blockIdx.x;            // 0..255
    int t = threadIdx.x;           // 0..255
    int lane = t >> 3;             // 0..31
    int j    = t & 7;              // bf16 pair index
    int s = f >> 4, nt = f & 15;
    int n = nt * 16 + (lane & 15);
    int kb = s * 32 + ((lane < 16) ? 0 : 8) + ((j < 4) ? 2 * j : 16 + 2 * (j - 4));
    pk[f * 512 + lane * 16 + 2 * j]     = f2bf_raw(Wk[(size_t)kb       * A_ + n]);
    pk[f * 512 + lane * 16 + 2 * j + 1] = f2bf_raw(Wk[(size_t)(kb + 1) * A_ + n]);
}

// ---------------------------------------------------------------------------
// Hot kernel: kf = keys_tile @ Wk via v_wmma_f32_16x16x32_bf16, fused
// tanh(qfeat + kf) . Wv epilogue -> scores.
// Block: (ktile, b), 256 threads = 8 waves, 128 key rows, 128 KB LDS tile.
// ---------------------------------------------------------------------------
__global__ void __launch_bounds__(THREADS)
k_scores(const float* __restrict__ keys, const unsigned char* __restrict__ mask,
         const float* __restrict__ qf, const unsigned short* __restrict__ wkp,
         const float* __restrict__ Wv, const float* __restrict__ bv,
         float* __restrict__ scores) {
    __shared__ unsigned short sk[TROWS * KS_];     // 128 KB bf16 key tile

    const int b = blockIdx.y;
    const int ktile = blockIdx.x;
    const int t = threadIdx.x;

    // cooperative load f32 tile -> convert -> bf16 LDS (row-major, 512/row)
    {
        const float4* src = (const float4*)(keys + ((size_t)b * KK + (size_t)ktile * TROWS) * KS_);
        for (int i = t; i < (TROWS * KS_ / 4); i += THREADS) {
            float4 v = src[i];
            unsigned int lo2 = (unsigned int)f2bf_raw(v.x) | ((unsigned int)f2bf_raw(v.y) << 16);
            unsigned int hi2 = (unsigned int)f2bf_raw(v.z) | ((unsigned int)f2bf_raw(v.w) << 16);
            ((uint2*)sk)[i] = make_uint2(lo2, hi2);
        }
    }
    __syncthreads();

    const int wave = t >> 5;
    const int lane = t & 31;
    const int lm   = lane & 15;
    const bool hihalf = lane >= 16;
    const unsigned short* apos = sk + (wave * 16 + lm) * KS_;   // A row for this lane
    const v16bf* bfr = (const v16bf*)wkp;

    float sc[8];
    #pragma unroll
    for (int r = 0; r < 8; ++r) sc[r] = 0.f;

    for (int ntc = 0; ntc < 4; ++ntc) {            // 4 chunks of 4 N-tiles
        v8f acc0 = {}, acc1 = {}, acc2 = {}, acc3 = {};
        for (int s = 0; s < 16; ++s) {             // K = 512 in steps of 32
            // A fragment: lanes<16 hold K {0..7,16..23}, lanes>=16 {8..15,24..31}
            int cb = s * 32 + (hihalf ? 8 : 0);
            v8bf alo = *(const v8bf*)(apos + cb);
            v8bf ahi = *(const v8bf*)(apos + cb + 16);
            v16bf a = __builtin_shufflevector(alo, ahi,
                        0,1,2,3,4,5,6,7,8,9,10,11,12,13,14,15);
            int fb = s * 16 + ntc * 4;
            v16bf b0 = bfr[(fb + 0) * 32 + lane];
            v16bf b1 = bfr[(fb + 1) * 32 + lane];
            v16bf b2 = bfr[(fb + 2) * 32 + lane];
            v16bf b3 = bfr[(fb + 3) * 32 + lane];
            acc0 = __builtin_amdgcn_wmma_f32_16x16x32_bf16(false, a, false, b0, (short)0, acc0, false, false);
            acc1 = __builtin_amdgcn_wmma_f32_16x16x32_bf16(false, a, false, b1, (short)0, acc1, false, false);
            acc2 = __builtin_amdgcn_wmma_f32_16x16x32_bf16(false, a, false, b2, (short)0, acc2, false, false);
            acc3 = __builtin_amdgcn_wmma_f32_16x16x32_bf16(false, a, false, b3, (short)0, acc3, false, false);
        }
        // fused epilogue: C layout -> lane holds N = nt*16 + lm for all 8 rows
        #pragma unroll
        for (int jj = 0; jj < 4; ++jj) {
            int n = (ntc * 4 + jj) * 16 + lm;
            float qv = qf[b * A_ + n];
            float wv = Wv[n];
            v8f acc = (jj == 0) ? acc0 : (jj == 1) ? acc1 : (jj == 2) ? acc2 : acc3;
            #pragma unroll
            for (int r = 0; r < 8; ++r)
                sc[r] += tanhf(acc[r] + qv) * wv;
        }
    }

    // reduce over the 16 lanes of each half (rows r in lanes 0-15, r+8 in 16-31)
    #pragma unroll
    for (int r = 0; r < 8; ++r) {
        float v = sc[r];
        v += __shfl_xor(v, 1);
        v += __shfl_xor(v, 2);
        v += __shfl_xor(v, 4);
        v += __shfl_xor(v, 8);
        sc[r] = v;
    }
    const float bv0 = bv[0];
    int rowbase = ktile * TROWS + wave * 16 + (hihalf ? 8 : 0);
    if (lm == 0) {
        #pragma unroll
        for (int r = 0; r < 8; ++r) {
            int k = rowbase + r;
            float sval = sc[r] + bv0;
            scores[b * KK + k] = mask[b * KK + k] ? sval : NEGV;
        }
    }
}

// ---------------------------------------------------------------------------
// In-place softmax over K per batch. 256 threads, 16 values/thread in regs.
// ---------------------------------------------------------------------------
__global__ void k_softmax(float* __restrict__ sw) {
    __shared__ float red[256];
    int b = blockIdx.x, t = threadIdx.x;
    float vals[16];
    float m = -3.4e38f;
    #pragma unroll
    for (int i = 0; i < 16; ++i) { vals[i] = sw[b * KK + i * 256 + t]; m = fmaxf(m, vals[i]); }
    red[t] = m; __syncthreads();
    for (int o = 128; o > 0; o >>= 1) { if (t < o) red[t] = fmaxf(red[t], red[t + o]); __syncthreads(); }
    m = red[0]; __syncthreads();
    float s = 0.f;
    #pragma unroll
    for (int i = 0; i < 16; ++i) { vals[i] = __expf(vals[i] - m); s += vals[i]; }
    red[t] = s; __syncthreads();
    for (int o = 128; o > 0; o >>= 1) { if (t < o) red[t] += red[t + o]; __syncthreads(); }
    float inv = 1.f / red[0];
    #pragma unroll
    for (int i = 0; i < 16; ++i) sw[b * KK + i * 256 + t] = vals[i] * inv;
}

// ---------------------------------------------------------------------------
// att partial: block (c, b) streams values[b, c*256 .. +256, :] as float4.
// 128 threads: thread t owns v = 4t..4t+3.
// ---------------------------------------------------------------------------
__global__ void k_attpart(const float* __restrict__ w, const float* __restrict__ values,
                          float* __restrict__ part) {
    __shared__ float lw[KK / KSPLIT];   // 256 weights
    int b = blockIdx.y, c = blockIdx.x, t = threadIdx.x;   // 128 threads
    for (int i = t; i < KK / KSPLIT; i += 128) lw[i] = w[b * KK + c * (KK / KSPLIT) + i];
    __syncthreads();
    const float4* vp = (const float4*)(values + ((size_t)b * KK + (size_t)c * (KK / KSPLIT)) * VD_) + t;
    float4 acc = make_float4(0.f, 0.f, 0.f, 0.f);
    for (int k = 0; k < KK / KSPLIT; ++k) {
        float4 v = vp[(size_t)k * (VD_ / 4)];
        float wk = lw[k];
        acc.x += wk * v.x; acc.y += wk * v.y; acc.z += wk * v.z; acc.w += wk * v.w;
    }
    ((float4*)part)[((size_t)b * KSPLIT + c) * (VD_ / 4) + t] = acc;
}

__global__ void k_attred(const float* __restrict__ part, float* __restrict__ out) {
    int b = blockIdx.x, t = threadIdx.x;   // 512 threads
    float acc = 0.f;
    #pragma unroll
    for (int c = 0; c < KSPLIT; ++c) acc += part[((size_t)b * KSPLIT + c) * VD_ + t];
    out[b * VD_ + t] = acc;
}

// ---------------------------------------------------------------------------
extern "C" void kernel_launch(void* const* d_in, const int* in_sizes, int n_in,
                              void* d_out, int out_size, void* d_ws, size_t ws_size,
                              hipStream_t stream) {
    const float* queries = (const float*)d_in[0];
    const float* keys    = (const float*)d_in[1];
    const float* values  = (const float*)d_in[2];
    const unsigned char* mask = (const unsigned char*)d_in[3];
    const float* Wq = (const float*)d_in[4];
    const float* bq = (const float*)d_in[5];
    const float* Wk = (const float*)d_in[6];
    const float* bk = (const float*)d_in[7];
    const float* Wv = (const float*)d_in[8];
    const float* bv = (const float*)d_in[9];
    float* out = (float*)d_out;

    char* ws = (char*)d_ws;
    float*          qfeat  = (float*)ws;                                  // 64 KB
    unsigned short* wkpack = (unsigned short*)(ws + 64 * 1024);           // 256 KB
    float*          scores = (float*)(ws + (64 + 256) * 1024);            // 1 MB (reused as weights)
    float*          part   = (float*)(ws + (64 + 256 + 1024) * 1024);     // 2 MB

    k_qfeat  <<<dim3(B_),             dim3(A_),      0, stream>>>(queries, Wq, bq, bk, qfeat);
    k_packWk <<<dim3(256),            dim3(256),     0, stream>>>(Wk, wkpack);
    k_scores <<<dim3(KK / TROWS, B_), dim3(THREADS), 0, stream>>>(keys, mask, qfeat, wkpack, Wv, bv, scores);
    k_softmax<<<dim3(B_),             dim3(256),     0, stream>>>(scores);
    k_attpart<<<dim3(KSPLIT, B_),     dim3(128),     0, stream>>>(scores, values, part);
    k_attred <<<dim3(B_),             dim3(VD_),     0, stream>>>(part, out);
}
